// ExtendedContextAttention_70609262346603
// MI455X (gfx1250) — compile-verified, matrix operand order checked by reading, not run on hardware
//
#include <hip/hip_runtime.h>
#include <hip/hip_bf16.h>
#include <math.h>

// ---------------------------------------------------------------------------
// Transformer block for MI455X (gfx1250, wave32, WMMA).
// All GEMMs + both attention einsums run on v_wmma_f32_16x16x32_bf16.
// ---------------------------------------------------------------------------

#define D_MODEL 1024
#define N_HEADS 16
#define HEAD_DIM 64
#define D_FF 4096
#define BATCH 2
#define SEQ 2048
#define NTOK (BATCH * SEQ)   // 4096

typedef __attribute__((ext_vector_type(16))) __bf16 v16bf;
typedef __attribute__((ext_vector_type(8)))  __bf16 v8bf;
typedef __attribute__((ext_vector_type(8)))  float  v8f;

static __device__ __forceinline__ v16bf ld_frag16(const __bf16* p) {
    // two contiguous 16B chunks per lane (ISA 16-bit A/B fragment layout)
    v8bf lo = *(const v8bf*)p;
    v8bf hi = *(const v8bf*)(p + 16);
    return __builtin_shufflevector(lo, hi, 0,1,2,3,4,5,6,7,8,9,10,11,12,13,14,15);
}

static __device__ __forceinline__ v8f wmma_bf16(v16bf a, v16bf b, v8f c) {
    return __builtin_amdgcn_wmma_f32_16x16x32_bf16(
        /*neg_a=*/false, a, /*neg_b=*/false, b,
        /*c_mod=*/(short)0, c, /*reuse_a=*/false, /*reuse_b=*/false);
}

static __device__ __forceinline__ float redmax16(float v) {
    v = fmaxf(v, __shfl_xor(v, 1, 32));
    v = fmaxf(v, __shfl_xor(v, 2, 32));
    v = fmaxf(v, __shfl_xor(v, 4, 32));
    v = fmaxf(v, __shfl_xor(v, 8, 32));
    return v;   // xor masks < 16: stays within 16-lane half of wave32
}
static __device__ __forceinline__ float redsum16(float v) {
    v += __shfl_xor(v, 1, 32);
    v += __shfl_xor(v, 2, 32);
    v += __shfl_xor(v, 4, 32);
    v += __shfl_xor(v, 8, 32);
    return v;
}

// ---------------------------------------------------------------------------
// Weight convert + transpose: wT[n*K + k] = bf16(w[k*N + n])
// ---------------------------------------------------------------------------
__global__ __launch_bounds__(256) void convT_kernel(const float* __restrict__ w,
                                                    __bf16* __restrict__ wT,
                                                    int K, int N) {
    long long idx = (long long)blockIdx.x * 256 + threadIdx.x;
    long long total = (long long)K * N;
    if (idx >= total) return;
    int k = (int)(idx / N);
    int n = (int)(idx % N);
    wT[(long long)n * K + k] = (__bf16)w[idx];
}

// ---------------------------------------------------------------------------
// LayerNorm over last dim (1024) -> bf16.  One 256-thread block per token.
// ---------------------------------------------------------------------------
__global__ __launch_bounds__(256) void ln_kernel(const float* __restrict__ x,
                                                 const float* __restrict__ g,
                                                 const float* __restrict__ b,
                                                 __bf16* __restrict__ out) {
    __shared__ float ssum[256], ssq[256];
    int tok = blockIdx.x;
    int tid = threadIdx.x;
    const float* row = x + (long long)tok * D_MODEL;
    float s = 0.f, q = 0.f;
    float v[4];
#pragma unroll
    for (int i = 0; i < 4; ++i) {
        v[i] = row[tid * 4 + i];
        s += v[i];
        q += v[i] * v[i];
    }
    ssum[tid] = s; ssq[tid] = q;
    __syncthreads();
    for (int off = 128; off > 0; off >>= 1) {
        if (tid < off) { ssum[tid] += ssum[tid + off]; ssq[tid] += ssq[tid + off]; }
        __syncthreads();
    }
    float mu  = ssum[0] * (1.0f / D_MODEL);
    float var = ssq[0] * (1.0f / D_MODEL) - mu * mu;
    float inv = rsqrtf(var + 1e-5f);
    __bf16* orow = out + (long long)tok * D_MODEL;
#pragma unroll
    for (int i = 0; i < 4; ++i) {
        int c = tid * 4 + i;
        orow[c] = (__bf16)((v[i] - mu) * inv * g[c] + b[c]);
    }
}

// ---------------------------------------------------------------------------
// Generic bf16 WMMA GEMM:  C[M,N] = A[M,K] @ Bt[N,K]^T  (+bias)(+gelu)(+res)
// 8 waves/block; wave tile 32x64; block tile 64x256; K-loop double buffered.
// MODE bits: 1=bias, 2=gelu(exact), 4=residual, 8=bf16 output
// ---------------------------------------------------------------------------
template <int MODE>
__global__ __launch_bounds__(256) void gemm_bf16_wmma(
    const __bf16* __restrict__ A, const __bf16* __restrict__ Bt,
    const float* __restrict__ bias, const float* __restrict__ res,
    void* __restrict__ Cout, int M, int N, int K)
{
    const int lane = threadIdx.x & 31;
    const int wid  = threadIdx.x >> 5;
    const int half = lane >> 4;
    const int l16  = lane & 15;
    const int rbase = blockIdx.y * 64 + (wid >> 2) * 32;
    const int cbase = blockIdx.x * 256 + (wid & 3) * 64;

    const __bf16* ap[2];
    const __bf16* bp[4];
#pragma unroll
    for (int mt = 0; mt < 2; ++mt)
        ap[mt] = A + (long long)(rbase + mt * 16 + l16) * K + half * 8;
#pragma unroll
    for (int nt = 0; nt < 4; ++nt)
        bp[nt] = Bt + (long long)(cbase + nt * 16 + l16) * K + half * 8;

    v8f acc[2][4];
    v8f z = {};
#pragma unroll
    for (int mt = 0; mt < 2; ++mt)
#pragma unroll
        for (int nt = 0; nt < 4; ++nt) acc[mt][nt] = z;

    // double-buffered fragment pipeline
    v16bf a_cur[2], b_cur[4];
#pragma unroll
    for (int mt = 0; mt < 2; ++mt) a_cur[mt] = ld_frag16(ap[mt]);
#pragma unroll
    for (int nt = 0; nt < 4; ++nt) b_cur[nt] = ld_frag16(bp[nt]);

    for (int kk = 0; kk < K; kk += 32) {
        v16bf a_nxt[2], b_nxt[4];
        int kn = kk + 32;
        if (kn < K) {
#pragma unroll
            for (int mt = 0; mt < 2; ++mt) a_nxt[mt] = ld_frag16(ap[mt] + kn);
#pragma unroll
            for (int nt = 0; nt < 4; ++nt) b_nxt[nt] = ld_frag16(bp[nt] + kn);
        }
#pragma unroll
        for (int mt = 0; mt < 2; ++mt)
#pragma unroll
            for (int nt = 0; nt < 4; ++nt)
                acc[mt][nt] = wmma_bf16(a_cur[mt], b_cur[nt], acc[mt][nt]);
        if (kn < K) {
#pragma unroll
            for (int mt = 0; mt < 2; ++mt) a_cur[mt] = a_nxt[mt];
#pragma unroll
            for (int nt = 0; nt < 4; ++nt) b_cur[nt] = b_nxt[nt];
        }
    }

#pragma unroll
    for (int mt = 0; mt < 2; ++mt)
#pragma unroll
        for (int nt = 0; nt < 4; ++nt)
#pragma unroll
            for (int j = 0; j < 8; ++j) {
                int row = rbase + mt * 16 + j + half * 8;
                int col = cbase + nt * 16 + l16;
                float v = acc[mt][nt][j];
                if (MODE & 1) v += bias[col];
                if (MODE & 2) v = 0.5f * v * (1.0f + erff(v * 0.70710678118654752f));
                if (MODE & 4) v += res[(long long)row * N + col];
                if (MODE & 8) ((__bf16*)Cout)[(long long)row * N + col] = (__bf16)v;
                else          ((float*)Cout)[(long long)row * N + col] = v;
            }
}

// ---------------------------------------------------------------------------
// RoPE + scatter: qkv f32 [NTOK, 3*1024] ->
//   Qbf [B,H,S,64], Kbf [B,H,S,64], Vt [B,H,64,S] (transposed for P@V frags)
// One thread per (b,h,s,pair d<32).
// ---------------------------------------------------------------------------
__global__ __launch_bounds__(256) void rope_kernel(const float* __restrict__ qkv,
                                                   __bf16* __restrict__ Q,
                                                   __bf16* __restrict__ Kb,
                                                   __bf16* __restrict__ Vt) {
    int idx = blockIdx.x * 256 + threadIdx.x;          // B*H*S*32 = 2M threads
    int d   = idx & 31;
    int s   = (idx >> 5) & (SEQ - 1);
    int h   = (idx >> 16) & (N_HEADS - 1);
    int b   = idx >> 20;
    if (b >= BATCH) return;

    long long rowbase = ((long long)b * SEQ + s) * (3 * D_MODEL);
    int co = h * HEAD_DIM + d;
    float q1 = qkv[rowbase + co],                q2 = qkv[rowbase + co + 32];
    float k1 = qkv[rowbase + D_MODEL + co],      k2 = qkv[rowbase + D_MODEL + co + 32];
    float v1 = qkv[rowbase + 2 * D_MODEL + co],  v2 = qkv[rowbase + 2 * D_MODEL + co + 32];

    float inv  = __expf(-logf(10000.0f) * (float)d * (1.0f / 32.0f));
    float ang  = (float)s * inv;
    float c = __cosf(ang), sn = __sinf(ang);

    long long qkbase = (((long long)(b * N_HEADS + h) * SEQ) + s) * HEAD_DIM;
    Q[qkbase + d]       = (__bf16)(q1 * c - q2 * sn);
    Q[qkbase + d + 32]  = (__bf16)(q2 * c + q1 * sn);
    Kb[qkbase + d]      = (__bf16)(k1 * c - k2 * sn);
    Kb[qkbase + d + 32] = (__bf16)(k2 * c + k1 * sn);

    long long vtb = ((long long)(b * N_HEADS + h) * HEAD_DIM);
    Vt[(vtb + d)      * SEQ + s] = (__bf16)v1;
    Vt[(vtb + d + 32) * SEQ + s] = (__bf16)v2;
}

// ---------------------------------------------------------------------------
// Flash attention, causal. One wave per 16-query tile; 8 waves/block.
// Scores via 2x chained WMMA (K=64); P@V via 4 WMMA per 32-key block.
// P redistributed C-layout -> A-layout through per-wave LDS (wave-sync).
// Output: attn_bf [NTOK, D_MODEL] bf16 (heads concatenated).
// ---------------------------------------------------------------------------
__global__ __launch_bounds__(256) void attn_kernel(const __bf16* __restrict__ Q,
                                                   const __bf16* __restrict__ Kb,
                                                   const __bf16* __restrict__ Vt,
                                                   __bf16* __restrict__ out) {
    __shared__ __align__(16) __bf16 plds[8][16 * 32];
    const int lane = threadIdx.x & 31;
    const int wid  = threadIdx.x >> 5;
    const int half = lane >> 4;
    const int l16  = lane & 15;

    const int gq    = blockIdx.x * 8 + wid;     // global query-tile id, 4096 total
    const int bh    = gq >> 7;                  // 128 tiles per (b,h)
    const int qt    = gq & 127;
    const int b     = bh >> 4;
    const int h     = bh & 15;
    const int qbase = qt * 16;
    const float scale = 0.125f;                 // 1/sqrt(64)

    const __bf16* Qp = Q  + (long long)bh * SEQ * HEAD_DIM;
    const __bf16* Kp = Kb + (long long)bh * SEQ * HEAD_DIM;
    const __bf16* Vp = Vt + (long long)bh * HEAD_DIM * SEQ;

    v16bf qa0 = ld_frag16(Qp + (long long)(qbase + l16) * HEAD_DIM + half * 8);
    v16bf qa1 = ld_frag16(Qp + (long long)(qbase + l16) * HEAD_DIM + 32 + half * 8);

    v8f z = {};
    v8f o0 = z, o1 = z, o2 = z, o3 = z;
    float mi[8], li[8];
#pragma unroll
    for (int j = 0; j < 8; ++j) { mi[j] = -1e30f; li[j] = 0.f; }

    for (int kb = 0; kb < qbase + 16; kb += 32) {
        // ---- S = Q @ K^T for keys [kb, kb+32) : batch all loads first ----
        const __bf16* k0 = Kp + (long long)(kb + l16) * HEAD_DIM + half * 8;
        const __bf16* k1 = Kp + (long long)(kb + 16 + l16) * HEAD_DIM + half * 8;
        v16bf k0a = ld_frag16(k0);
        v16bf k0b = ld_frag16(k0 + 32);
        v16bf k1a = ld_frag16(k1);
        v16bf k1b = ld_frag16(k1 + 32);
        // interleave the two accumulation chains (hides WMMA RAW hazard)
        v8f s0 = wmma_bf16(qa0, k0a, z);
        v8f s1 = wmma_bf16(qa0, k1a, z);
        s0 = wmma_bf16(qa1, k0b, s0);
        s1 = wmma_bf16(qa1, k1b, s1);

        // ---- online softmax, causal mask (fully unrolled: static regs) ----
        int nk0 = kb + l16, nk1 = kb + 16 + l16;
#pragma unroll
        for (int j = 0; j < 8; ++j) {
            int rowq = qbase + j + half * 8;
            float a0 = s0[j] * scale; if (nk0 > rowq) a0 = -1e9f;
            float a1 = s1[j] * scale; if (nk1 > rowq) a1 = -1e9f;
            float rm = redmax16(fmaxf(a0, a1));
            rm = fmaxf(rm, mi[j]);
            float f  = __expf(mi[j] - rm);
            float p0 = __expf(a0 - rm);
            float p1 = __expf(a1 - rm);
            li[j] = li[j] * f + redsum16(p0 + p1);
            mi[j] = rm;
            o0[j] *= f; o1[j] *= f; o2[j] *= f; o3[j] *= f;
            int prow = (j + half * 8) * 32;
            plds[wid][prow + l16]      = (__bf16)p0;
            plds[wid][prow + 16 + l16] = (__bf16)p1;
        }
        asm volatile("s_wait_dscnt 0" ::: "memory");   // wave-sync LDS RAW

        // ---- gather P in A-fragment layout ----
        v16bf pa = ld_frag16(&plds[wid][l16 * 32 + half * 8]);

        // ---- O += P @ V : batch loads, then 4 independent WMMAs ----
        long long vo = (long long)kb + half * 8;
        v16bf vf0 = ld_frag16(Vp + (long long)(l16)      * SEQ + vo);
        v16bf vf1 = ld_frag16(Vp + (long long)(16 + l16) * SEQ + vo);
        v16bf vf2 = ld_frag16(Vp + (long long)(32 + l16) * SEQ + vo);
        v16bf vf3 = ld_frag16(Vp + (long long)(48 + l16) * SEQ + vo);
        o0 = wmma_bf16(pa, vf0, o0);
        o1 = wmma_bf16(pa, vf1, o1);
        o2 = wmma_bf16(pa, vf2, o2);
        o3 = wmma_bf16(pa, vf3, o3);
    }

    // ---- normalize + store bf16 [tok, h*64 + feat] ----
#pragma unroll
    for (int j = 0; j < 8; ++j) {
        float inv = 1.0f / li[j];
        long long tok = (long long)b * SEQ + qbase + j + half * 8;
        __bf16* orow = out + tok * D_MODEL + h * HEAD_DIM;
        orow[l16]      = (__bf16)(o0[j] * inv);
        orow[16 + l16] = (__bf16)(o1[j] * inv);
        orow[32 + l16] = (__bf16)(o2[j] * inv);
        orow[48 + l16] = (__bf16)(o3[j] * inv);
    }
}

// ---------------------------------------------------------------------------
// Host orchestration
// ---------------------------------------------------------------------------
extern "C" void kernel_launch(void* const* d_in, const int* in_sizes, int n_in,
                              void* d_out, int out_size, void* d_ws, size_t ws_size,
                              hipStream_t stream) {
    const float* x     = (const float*)d_in[0];
    const float* w_qkv = (const float*)d_in[1];
    const float* w_o   = (const float*)d_in[2];
    const float* ln1_g = (const float*)d_in[3];
    const float* ln1_b = (const float*)d_in[4];
    const float* ln2_g = (const float*)d_in[5];
    const float* ln2_b = (const float*)d_in[6];
    const float* w1    = (const float*)d_in[7];
    const float* b1    = (const float*)d_in[8];
    const float* w2    = (const float*)d_in[9];
    const float* b2    = (const float*)d_in[10];

    char* p = (char*)d_ws;
    auto alloc = [&](size_t bytes) { void* r = p; p += (bytes + 255) & ~(size_t)255; return r; };

    __bf16* wqkvT  = (__bf16*)alloc((size_t)D_MODEL * 3 * D_MODEL * 2);
    __bf16* woT    = (__bf16*)alloc((size_t)D_MODEL * D_MODEL * 2);
    __bf16* w1T    = (__bf16*)alloc((size_t)D_MODEL * D_FF * 2);
    __bf16* w2T    = (__bf16*)alloc((size_t)D_FF * D_MODEL * 2);
    __bf16* h1     = (__bf16*)alloc((size_t)NTOK * D_MODEL * 2);
    float*  qkv    = (float*)alloc((size_t)NTOK * 3 * D_MODEL * 4);
    __bf16* Qbf    = (__bf16*)alloc((size_t)NTOK * D_MODEL * 2);
    __bf16* Kbf    = (__bf16*)alloc((size_t)NTOK * D_MODEL * 2);
    __bf16* Vtb    = (__bf16*)alloc((size_t)NTOK * D_MODEL * 2);
    __bf16* attnbf = (__bf16*)alloc((size_t)NTOK * D_MODEL * 2);
    float*  x2     = (float*)alloc((size_t)NTOK * D_MODEL * 4);
    __bf16* h2     = (__bf16*)alloc((size_t)NTOK * D_MODEL * 2);
    __bf16* gbuf   = (__bf16*)alloc((size_t)NTOK * D_FF * 2);

    // weight convert+transpose (f32 -> bf16, [K,N] -> [N,K])
    convT_kernel<<<(D_MODEL * 3 * D_MODEL + 255) / 256, 256, 0, stream>>>(w_qkv, wqkvT, D_MODEL, 3 * D_MODEL);
    convT_kernel<<<(D_MODEL * D_MODEL + 255) / 256, 256, 0, stream>>>(w_o, woT, D_MODEL, D_MODEL);
    convT_kernel<<<(D_MODEL * D_FF + 255) / 256, 256, 0, stream>>>(w1, w1T, D_MODEL, D_FF);
    convT_kernel<<<(D_FF * D_MODEL + 255) / 256, 256, 0, stream>>>(w2, w2T, D_FF, D_MODEL);

    // h1 = LN1(x)
    ln_kernel<<<NTOK, 256, 0, stream>>>(x, ln1_g, ln1_b, h1);

    // qkv = h1 @ w_qkv
    gemm_bf16_wmma<0><<<dim3(3 * D_MODEL / 256, NTOK / 64), 256, 0, stream>>>(
        h1, wqkvT, nullptr, nullptr, qkv, NTOK, 3 * D_MODEL, D_MODEL);

    // RoPE + per-head scatter
    rope_kernel<<<(BATCH * N_HEADS * SEQ * 32) / 256, 256, 0, stream>>>(qkv, Qbf, Kbf, Vtb);

    // causal flash attention
    attn_kernel<<<(BATCH * N_HEADS * (SEQ / 16)) / 8, 256, 0, stream>>>(Qbf, Kbf, Vtb, attnbf);

    // x2 = x + attn @ w_o
    gemm_bf16_wmma<4><<<dim3(D_MODEL / 256, NTOK / 64), 256, 0, stream>>>(
        attnbf, woT, nullptr, x, x2, NTOK, D_MODEL, D_MODEL);

    // h2 = LN2(x2)
    ln_kernel<<<NTOK, 256, 0, stream>>>(x2, ln2_g, ln2_b, h2);

    // g = gelu(h2 @ w1 + b1)
    gemm_bf16_wmma<1 | 2 | 8><<<dim3(D_FF / 256, NTOK / 64), 256, 0, stream>>>(
        h2, w1T, b1, nullptr, gbuf, NTOK, D_FF, D_MODEL);

    // out = x2 + g @ w2 + b2
    gemm_bf16_wmma<1 | 4><<<dim3(D_MODEL / 256, NTOK / 64), 256, 0, stream>>>(
        gbuf, w2T, b2, x2, d_out, NTOK, D_MODEL, D_FF);
}
